// SeamlessM4Tv2ConformerSelfAttention_52278341927425
// MI455X (gfx1250) — compile-verified, matrix operand order checked by reading, not run on hardware
//
#include <hip/hip_runtime.h>
#include <hip/hip_bf16.h>
#if defined(__has_include)
#if __has_include(<hip/amd_detail/amd_gfx1250_TDM.h>)
#include <hip/amd_detail/amd_gfx1250_TDM.h>
#endif
#endif

// ---------------------------------------------------------------------------
// SeamlessM4Tv2 conformer self-attention, CDNA5 (gfx1250) WMMA implementation.
// B=8, S=1024, HID=1024, NH=16, DH=64, LEFT_MAX=64, RIGHT_MAX=8.
// All matrix math via v_wmma_f32_16x16x32_f16; all LDS tile staging via the
// Tensor Data Mover with double-buffering (DMA overlapped with WMMA).
// ---------------------------------------------------------------------------

typedef __attribute__((ext_vector_type(16))) _Float16     v16h;
typedef __attribute__((ext_vector_type(8)))  _Float16     v8h;
typedef __attribute__((ext_vector_type(8)))  float        v8f;
typedef __attribute__((ext_vector_type(4)))  unsigned int u32x4;
typedef __attribute__((ext_vector_type(8)))  int          i32x8;
typedef __attribute__((ext_vector_type(4)))  int          i32x4;

#define SEQ   1024
#define NH    16
#define DH    64
#define HID   1024
#define BSZ   8
#define MTOT  (BSZ * SEQ)      // 8192 rows for the dense GEMMs
#define RELW  80               // 73 distances padded to 80
#define NROWS (BSZ * NH * SEQ) // 131072 q rows

// D = A(16x32 f16) x B(32x16 f16) + C(f32)
__device__ __forceinline__ v8f wmma_f16(v16h a, v16h b, v8f c) {
  return __builtin_amdgcn_wmma_f32_16x16x32_f16(
      /*neg_a=*/false, a, /*neg_b=*/false, b,
      /*c_mod=*/(short)0, c, /*reuse_a=*/false, /*reuse_b=*/false);
}

// Load a 16x32 f16 WMMA fragment from an LDS tile (row-major A tile, or a
// column-major B tile; same per-lane pattern). ld = leading dim in halves.
// ISA layout: lanes 0-15: row=lane, K0..7 halves 0..7, K16..23 halves 8..15;
// lanes 16-31: row=lane&15, K8..15 and K24..31.
__device__ __forceinline__ v16h load_frag(const _Float16* base, int ld) {
  const int lane = threadIdx.x & 31;
  const int r    = lane & 15;
  const int k0   = (lane >> 4) << 3;             // 0 or 8
  v8h lo = *reinterpret_cast<const v8h*>(base + r * ld + k0);
  v8h hi = *reinterpret_cast<const v8h*>(base + r * ld + k0 + 16);
  v16h f;
#pragma unroll
  for (int i = 0; i < 8; ++i) { f[i] = lo[i]; f[i + 8] = hi[i]; }
  return f;
}

// ---------------------- Tensor Data Mover tile loads -----------------------
// Loads a rows x 64 f16 tile (global row stride = rowStrideElems halves) into
// LDS with pad_enable inserting 4 DWORDs (8 halves) after every 32 DWORDs
// (64 halves = one row) -> effective LDS row stride of 72 halves.
// D# bit layout per CDNA5 ISA ch.8 (group0 128b, group1 256b; groups 2/3 zero
// for 2D tiles). Issued by one wave; completion via s_wait_tensorcnt.
__device__ __forceinline__ void tdm_load_tile_f16(const _Float16* gptr,
                                                  unsigned int ldsByteOff,
                                                  unsigned int rowStrideElems,
                                                  unsigned int rows) {
  unsigned long long ga = (unsigned long long)(size_t)gptr;
  u32x4 g0;
  g0[0] = 1u;                                          // count=1 (valid), rest 0
  g0[1] = ldsByteOff;                                  // lds_addr (bytes)
  g0[2] = (unsigned int)(ga & 0xFFFFFFFFu);            // global_addr[31:0]
  g0[3] = (unsigned int)((ga >> 32) & 0x01FFFFFFu)     // global_addr[56:32]
        | 0x80000000u;                                 // type = 2 ("image")
  unsigned long long d1s = (unsigned long long)rowStrideElems * 64ull;
  i32x8 g1;
  g1[0] = (1 << 16)                                    // data_size = 2 bytes
        | (1 << 20)                                    // pad_enable
        | (4 << 22)                                    // pad_interval: 32 DWORDs
        | (3 << 25);                                   // pad_amount: 4 DWORDs
  g1[1] = (int)(64u << 16);                            // tensor_dim0 = 64 (lo16)
  g1[2] = (int)(rows << 16);                           // tensor_dim1 (lo16)
  g1[3] = (int)(64u << 16);                            // tile_dim0 = 64
  g1[4] = (int)rows;                                   // tile_dim1 = rows
  g1[5] = (int)rowStrideElems;                         // tensor_dim0_stride[31:0]
  g1[6] = (int)(((unsigned int)(d1s & 0xFFFFull)) << 16); // dim1_stride[15:0]
  g1[7] = (int)(unsigned int)(d1s >> 16);              // dim1_stride[47:16]
  i32x4 z4 = {0, 0, 0, 0};
  i32x8 z8 = {0, 0, 0, 0, 0, 0, 0, 0};
  __builtin_amdgcn_tensor_load_to_lds(g0, g1, z4, z4, z8, 0);
}

// ------------------------- small conversion kernels -------------------------

__global__ void cvt_f32_to_f16(const float* __restrict__ src,
                               _Float16* __restrict__ dst, int n) {
  int i = blockIdx.x * blockDim.x + threadIdx.x;
  if (i < n) dst[i] = (_Float16)src[i];
}

// Wt[n*HID + k] = (f16) W[k*HID + n]   (column-major f16 weights)
__global__ void transpose_w_f16(const float* __restrict__ W,
                                _Float16* __restrict__ Wt) {
  int idx = blockIdx.x * blockDim.x + threadIdx.x;   // HID*HID threads
  int n = idx >> 10, k = idx & (HID - 1);
  Wt[idx] = (_Float16)W[k * HID + n];
}

// ------------------------------ WMMA GEMM ----------------------------------
// C[M,N] = A[M,K](f16,row-major) * Bt[N,K](f16,col-major of B) + bias[N]
// 128x128 block tile, K-step 64, TDM double-buffered LDS staging. 256 threads
// = 8 waves in a 4x2 grid; each wave computes 32x64 of C (8 f32 accumulators,
// 16 WMMA per K-step).
// MODE 0: f32 row-major [M,N]  (final projection -> d_out)
// MODE 1: f16 [B,nh,S,dh]      (Q and K)
// MODE 2: f16 [B,nh,dh,S]      (V transposed)

template <int MODE>
__device__ __forceinline__ void store_out(void* outp, int m, int n, float v) {
  const int b = m >> 10, s = m & (SEQ - 1);
  const int h = n >> 6,  d = n & (DH - 1);
  if constexpr (MODE == 0) {
    reinterpret_cast<float*>(outp)[(size_t)m * HID + n] = v;
  } else if constexpr (MODE == 1) {
    reinterpret_cast<_Float16*>(outp)
        [(((size_t)(b * NH + h) * SEQ) + s) * DH + d] = (_Float16)v;
  } else {
    reinterpret_cast<_Float16*>(outp)
        [(((size_t)(b * NH + h) * DH) + d) * SEQ + s] = (_Float16)v;
  }
}

template <int MODE>
__global__ void __launch_bounds__(256)
gemm_wmma_128x128(const _Float16* __restrict__ A, const _Float16* __restrict__ Bt,
                  const float* __restrict__ bias, void* __restrict__ outp,
                  int M, int N, int K) {
  constexpr int LDT = 72;                       // 64 + 8 halves (144B rows)
  __shared__ _Float16 As[2][128 * LDT];
  __shared__ _Float16 Bs[2][128 * LDT];

  const int tid  = threadIdx.x;
  const int lane = tid & 31;
  const int w    = tid >> 5;
  const int wr   = w >> 1;                      // 0..3  (32-row group)
  const int wc   = w & 1;                       // 0..1  (64-col group)
  const int ntn  = N >> 7;
  const int m0   = (blockIdx.x / ntn) * 128;
  const int n0   = (blockIdx.x % ntn) * 128;

  const _Float16* Abase = A  + (size_t)m0 * K;
  const _Float16* Bbase = Bt + (size_t)n0 * K;

  v8f acc[2][4];
#pragma unroll
  for (int g = 0; g < 2; ++g)
#pragma unroll
    for (int j = 0; j < 4; ++j) acc[g][j] = (v8f){};

  // preload first K-tiles via TDM
  if (w == 0) {
    tdm_load_tile_f16(Abase, (unsigned int)(size_t)&As[0][0], K, 128);
    tdm_load_tile_f16(Bbase, (unsigned int)(size_t)&Bs[0][0], K, 128);
  }

  const int NK = K >> 6;
  int p = 0;
  for (int kt = 0; kt < NK; ++kt) {
    if (w == 0) {
      if (kt + 1 < NK) {
        // DMA next K-tiles into the other buffer, then wait for current pair
        // (TENSORcnt completes in order: <=2 outstanding => current landed).
        tdm_load_tile_f16(Abase + (kt + 1) * 64,
                          (unsigned int)(size_t)&As[p ^ 1][0], K, 128);
        tdm_load_tile_f16(Bbase + (kt + 1) * 64,
                          (unsigned int)(size_t)&Bs[p ^ 1][0], K, 128);
        __builtin_amdgcn_s_wait_tensorcnt(2);
      } else {
        __builtin_amdgcn_s_wait_tensorcnt(0);
      }
    }
    __syncthreads();                            // publish buffer p

#pragma unroll
    for (int kk = 0; kk < 64; kk += 32) {
      v16h a0 = load_frag(&As[p][(wr * 32)      * LDT + kk], LDT);
      v16h a1 = load_frag(&As[p][(wr * 32 + 16) * LDT + kk], LDT);
      v16h b0 = load_frag(&Bs[p][(wc * 64)      * LDT + kk], LDT);
      v16h b1 = load_frag(&Bs[p][(wc * 64 + 16) * LDT + kk], LDT);
      v16h b2 = load_frag(&Bs[p][(wc * 64 + 32) * LDT + kk], LDT);
      v16h b3 = load_frag(&Bs[p][(wc * 64 + 48) * LDT + kk], LDT);
      acc[0][0] = wmma_f16(a0, b0, acc[0][0]);
      acc[0][1] = wmma_f16(a0, b1, acc[0][1]);
      acc[0][2] = wmma_f16(a0, b2, acc[0][2]);
      acc[0][3] = wmma_f16(a0, b3, acc[0][3]);
      acc[1][0] = wmma_f16(a1, b0, acc[1][0]);
      acc[1][1] = wmma_f16(a1, b1, acc[1][1]);
      acc[1][2] = wmma_f16(a1, b2, acc[1][2]);
      acc[1][3] = wmma_f16(a1, b3, acc[1][3]);
    }
    __syncthreads();                            // retire buffer p before reuse
    p ^= 1;
  }

  const int colL = lane & 15;
  const int rOff = (lane >> 4) << 3;
#pragma unroll
  for (int g = 0; g < 2; ++g) {
#pragma unroll
    for (int j = 0; j < 4; ++j) {
#pragma unroll
      for (int r = 0; r < 8; ++r) {
        const int mg = m0 + wr * 32 + g * 16 + rOff + r;
        const int ng = n0 + wc * 64 + j * 16 + colL;
        store_out<MODE>(outp, mg, ng, acc[g][j][r] + bias[ng]);
      }
    }
  }
}

// --------------------------- relative-bias table ---------------------------
// relb[row, dist] = (1/8) * dot(Q[row,:], dist_emb[dist,:]),  dist in [0,73)
__global__ void relbias_kernel(const _Float16* __restrict__ Q,
                               const float* __restrict__ dist_emb,
                               float* __restrict__ relb) {
  size_t gid = (size_t)blockIdx.x * blockDim.x + threadIdx.x;
  int    dist = (int)(gid % RELW);
  size_t row  = gid / RELW;
  if (row >= (size_t)NROWS) return;
  float v = 0.f;
  if (dist < 73) {
    const _Float16* qr = Q + row * DH;
    const float*    de = dist_emb + (size_t)dist * DH;
#pragma unroll 8
    for (int i = 0; i < DH; ++i) v += (float)qr[i] * de[i];
    v *= 0.125f;                                  // 1/sqrt(64)
  }
  relb[row * RELW + dist] = v;
}

// ------------------------------ flash attention ----------------------------
// One block = one (b, h, 64-query-row tile). 256 threads = 8 waves (4x2).
// K/V tiles DMA'd by the TDM into ping-pong LDS buffers: wave 0 issues the
// NEXT tile pair, waits tensorcnt<=2 (in-order completion => current pair
// done), then the workgroup barrier publishes. DMA overlaps WMMA + softmax.
__global__ void __launch_bounds__(256)
attn_kernel(const _Float16* __restrict__ Q, const _Float16* __restrict__ Kmat,
            const _Float16* __restrict__ Vt, const float* __restrict__ relb,
            _Float16* __restrict__ AO) {
  constexpr int LD = 72;                          // 64+8 halves, 144B rows
  __shared__ _Float16 Qs[64 * LD];
  __shared__ _Float16 Ks[2][64 * LD];
  __shared__ _Float16 Vs[2][64 * LD];             // Vs[.][d][key]
  __shared__ _Float16 Ps[64 * LD];
  __shared__ float    Ss[64 * 64];
  __shared__ float    Rb[64 * RELW];
  __shared__ float    rowm[64], rowl[64], rowsc[64];

  const int tid  = threadIdx.x;
  const int lane = tid & 31;
  const int w    = tid >> 5;
  const int wr   = w >> 1, wc = w & 1;
  const int mt   = blockIdx.x & 15;
  const int h    = (blockIdx.x >> 4) & 15;
  const int b    = blockIdx.x >> 8;
  const size_t headOff = (size_t)(b * NH + h) * SEQ * DH;
  const float inv_sqrt_d = 0.125f;

  const _Float16* Kbase = Kmat + headOff;
  const _Float16* Vbase = Vt + headOff;

  // TDM prologue: Q tile + first K/V tiles (3 ops outstanding on wave 0)
  if (w == 0) {
    tdm_load_tile_f16(Q + headOff + (size_t)(mt * 64) * DH,
                      (unsigned int)(size_t)&Qs[0], DH, 64);
    tdm_load_tile_f16(Kbase, (unsigned int)(size_t)&Ks[0][0], DH, 64);
    tdm_load_tile_f16(Vbase, (unsigned int)(size_t)&Vs[0][0], SEQ, 64);
  }
  // stage this tile's bias rows (f32, manual)
  for (int t = tid; t < 64 * (RELW / 4); t += 256) {
    int r = t / (RELW / 4), sgf = t % (RELW / 4);
    *reinterpret_cast<float4*>(&Rb[r * RELW + sgf * 4]) =
        *reinterpret_cast<const float4*>(
            relb + ((size_t)(b * NH + h) * SEQ + mt * 64 + r) * RELW + sgf * 4);
  }
  if (tid < 64) { rowm[tid] = -1e30f; rowl[tid] = 0.f; }

  v8f o0 = {}; v8f o1 = {};
  __syncthreads();                                // publish Rb / row stats

  const int colL = lane & 15;
  const int rOff = (lane >> 4) << 3;
  constexpr int NB = SEQ / 64;

  for (int kb = 0; kb < NB; ++kb) {
    const int cur = kb & 1;
    if (w == 0) {
      if (kb + 1 < NB) {
        tdm_load_tile_f16(Kbase + (size_t)((kb + 1) * 64) * DH,
                          (unsigned int)(size_t)&Ks[cur ^ 1][0], DH, 64);
        tdm_load_tile_f16(Vbase + (kb + 1) * 64,
                          (unsigned int)(size_t)&Vs[cur ^ 1][0], SEQ, 64);
        __builtin_amdgcn_s_wait_tensorcnt(2);     // current tiles (and Q) landed
      } else {
        __builtin_amdgcn_s_wait_tensorcnt(0);
      }
    }
    __syncthreads();                              // publish Ks/Vs[cur] (+Qs)

    // S = Q * K^T  (per wave: 16 rows x 32 keys, K-dim 64 -> 2 steps)
    v8f s0 = {}; v8f s1 = {};
#pragma unroll
    for (int kk = 0; kk < DH; kk += 32) {
      v16h a  = load_frag(&Qs[(wr * 16) * LD + kk], LD);
      v16h b0 = load_frag(&Ks[cur][(wc * 32) * LD + kk], LD);
      v16h b1 = load_frag(&Ks[cur][(wc * 32 + 16) * LD + kk], LD);
      s0 = wmma_f16(a, b0, s0);
      s1 = wmma_f16(a, b1, s1);
    }
#pragma unroll
    for (int r = 0; r < 8; ++r) {
      const int row  = wr * 16 + rOff + r;
      const int rowG = mt * 64 + row;
      const int k0   = wc * 32 + colL;
      const int kg0  = kb * 64 + k0;
      const int kg1  = kg0 + 16;
      const int d0   = min(max(kg0 - rowG, -64), 8) + 64;
      const int d1   = min(max(kg1 - rowG, -64), 8) + 64;
      Ss[row * 64 + k0]      = s0[r] * inv_sqrt_d + Rb[row * RELW + d0];
      Ss[row * 64 + k0 + 16] = s1[r] * inv_sqrt_d + Rb[row * RELW + d1];
    }
    __syncthreads();

    // online softmax: 4 lanes per row, 16 cols each
    {
      const int row = tid >> 2, q = tid & 3;
      float sv[16];
      float mloc = -1e30f;
#pragma unroll
      for (int c = 0; c < 16; ++c) {
        sv[c] = Ss[row * 64 + q * 16 + c];
        mloc  = fmaxf(mloc, sv[c]);
      }
      mloc = fmaxf(mloc, __shfl_xor(mloc, 1, 32));
      mloc = fmaxf(mloc, __shfl_xor(mloc, 2, 32));
      const float mold = rowm[row];
      const float mnew = fmaxf(mold, mloc);
      const float sc   = __expf(mold - mnew);
      float sum = 0.f;
#pragma unroll
      for (int c = 0; c < 16; ++c) {
        float pv = __expf(sv[c] - mnew);
        Ps[row * LD + q * 16 + c] = (_Float16)pv;
        sum += pv;
      }
      sum += __shfl_xor(sum, 1, 32);
      sum += __shfl_xor(sum, 2, 32);
      if (q == 0) { rowl[row] = rowl[row] * sc + sum; rowm[row] = mnew; rowsc[row] = sc; }
    }
    __syncthreads();

    // O = O*scale + P*V   (per wave: 16 rows x 32 dh cols)
#pragma unroll
    for (int r = 0; r < 8; ++r) {
      const float sc = rowsc[wr * 16 + rOff + r];
      o0[r] *= sc; o1[r] *= sc;
    }
#pragma unroll
    for (int kk = 0; kk < 64; kk += 32) {
      v16h a  = load_frag(&Ps[(wr * 16) * LD + kk], LD);
      v16h b0 = load_frag(&Vs[cur][(wc * 32) * LD + kk], LD);
      v16h b1 = load_frag(&Vs[cur][(wc * 32 + 16) * LD + kk], LD);
      o0 = wmma_f16(a, b0, o0);
      o1 = wmma_f16(a, b1, o1);
    }
    __syncthreads();                              // retire Ks/Vs[cur] before DMA reuse
  }

  // normalize and write AO as [B*S, HID] f16 for the output projection
#pragma unroll
  for (int r = 0; r < 8; ++r) {
    const int row  = wr * 16 + rOff + r;
    const float il = 1.0f / rowl[row];
    const int   mg = b * SEQ + mt * 64 + row;
    const int   d0 = wc * 32 + colL;
    AO[(size_t)mg * HID + h * DH + d0]      = (_Float16)(o0[r] * il);
    AO[(size_t)mg * HID + h * DH + d0 + 16] = (_Float16)(o1[r] * il);
  }
}

// ------------------------------- launcher ----------------------------------

extern "C" void kernel_launch(void* const* d_in, const int* in_sizes, int n_in,
                              void* d_out, int out_size, void* d_ws, size_t ws_size,
                              hipStream_t stream) {
  (void)in_sizes; (void)n_in; (void)out_size; (void)ws_size;
  const float* hs  = (const float*)d_in[0];
  const float* Wq  = (const float*)d_in[1];
  const float* bq  = (const float*)d_in[2];
  const float* Wk  = (const float*)d_in[3];
  const float* bk  = (const float*)d_in[4];
  const float* Wv  = (const float*)d_in[5];
  const float* bv  = (const float*)d_in[6];
  const float* Wo  = (const float*)d_in[7];
  const float* bo  = (const float*)d_in[8];
  const float* de  = (const float*)d_in[9];

  char*  ws  = (char*)d_ws;
  size_t off = 0;
  auto alloc = [&](size_t bytes) -> void* {
    void* p = ws + off;
    off = (off + bytes + 255) & ~(size_t)255;
    return p;
  };

  _Float16* Xh   = (_Float16*)alloc((size_t)MTOT * HID * 2);      // reused as AO
  _Float16* WqT  = (_Float16*)alloc((size_t)HID * HID * 2);
  _Float16* WkT  = (_Float16*)alloc((size_t)HID * HID * 2);
  _Float16* WvT  = (_Float16*)alloc((size_t)HID * HID * 2);
  _Float16* WoT  = (_Float16*)alloc((size_t)HID * HID * 2);
  _Float16* Qh   = (_Float16*)alloc((size_t)NROWS * DH * 2);
  _Float16* Kh   = (_Float16*)alloc((size_t)NROWS * DH * 2);
  _Float16* Vth  = (_Float16*)alloc((size_t)NROWS * DH * 2);
  float*    relb = (float*)   alloc((size_t)NROWS * RELW * 4);
  _Float16* AO   = Xh;  // hidden f16 no longer needed after QKV projections

  // 1) conversions
  cvt_f32_to_f16<<<(MTOT * HID) / 256, 256, 0, stream>>>(hs, Xh, MTOT * HID);
  transpose_w_f16<<<(HID * HID) / 256, 256, 0, stream>>>(Wq, WqT);
  transpose_w_f16<<<(HID * HID) / 256, 256, 0, stream>>>(Wk, WkT);
  transpose_w_f16<<<(HID * HID) / 256, 256, 0, stream>>>(Wv, WvT);
  transpose_w_f16<<<(HID * HID) / 256, 256, 0, stream>>>(Wo, WoT);

  // 2) QKV projections (WMMA + TDM double-buffered staging)
  const int gemm_blocks = (MTOT / 128) * (HID / 128);   // 64*8 = 512
  gemm_wmma_128x128<1><<<gemm_blocks, 256, 0, stream>>>(Xh, WqT, bq, Qh,  MTOT, HID, HID);
  gemm_wmma_128x128<1><<<gemm_blocks, 256, 0, stream>>>(Xh, WkT, bk, Kh,  MTOT, HID, HID);
  gemm_wmma_128x128<2><<<gemm_blocks, 256, 0, stream>>>(Xh, WvT, bv, Vth, MTOT, HID, HID);

  // 3) relative-position bias table
  relbias_kernel<<<((size_t)NROWS * RELW) / 256, 256, 0, stream>>>(Qh, de, relb);

  // 4) flash attention (WMMA + ping-pong TDM staging), writes AO [B*S,HID] f16
  attn_kernel<<<BSZ * NH * (SEQ / 64), 256, 0, stream>>>(Qh, Kh, Vth, relb, AO);

  // 5) output projection (WMMA) -> f32 d_out
  gemm_wmma_128x128<0><<<gemm_blocks, 256, 0, stream>>>(AO, WoT, bo, d_out, MTOT, HID, HID);
}
